// LSTMLayer_45088566673865
// MI455X (gfx1250) — compile-verified, matrix operand order checked by reading, not run on hardware
//
#include <hip/hip_runtime.h>

// ---------------------------------------------------------------------------
// LSTM layer for MI455X (gfx1250, wave32, WMMA).
//   S=512, B=64, I=H=1024, G=4H=4096.
// Phase 1: gates_x = x @ W_ih^T + (b_ih + b_hh)   -- bf16 WMMA GEMM,
//          64x64 register tile per wave (16 WMMAs / 8 fragment loads per k).
//          gates_x stored in WMMA C-fragment tile order (32 lanes x 8 floats
//          per 16x16 tile) so both store and reload are contiguous b128s.
// Phase 2: 512 sequential steps: gates = gates_x[t] + h @ W_hh^T, cell update.
// Weights pre-swizzled to WMMA-fragment order; h kept in bf16 ping-pong bufs.
// ---------------------------------------------------------------------------

typedef __attribute__((ext_vector_type(16))) __bf16 v16bf;
typedef __attribute__((ext_vector_type(8)))  __bf16 v8bf;
typedef __attribute__((ext_vector_type(4)))  __bf16 v4bf;
typedef __attribute__((ext_vector_type(8)))  float  v8f;

#define S_LEN 512
#define B_DIM 64
#define I_DIM 1024
#define H_DIM 1024
#define G_DIM 4096               // 4*H
#define SB    (S_LEN * B_DIM)    // 32768 rows of x
#define KT_I  (I_DIM / 32)       // 32 k-tiles for input GEMM
#define KT_H  (H_DIM / 32)       // 32 k-tiles for recurrent GEMM
#define NT_G  (G_DIM / 16)       // 256 column tiles of the gate matrix

union AFrag { v16bf v; v8bf h[2]; };

__device__ __forceinline__ float sigf(float x) {
  return 1.0f / (1.0f + __expf(-x));
}
__device__ __forceinline__ float tanhfast(float x) {
  float e = __expf(-2.0f * x);
  return (1.0f - e) / (1.0f + e);
}

// ---- weight pre-swizzle: packed[((nt*KT + kt)*32 + lane)*16 + j] = bf16(W[n,k])
// n = nt*16 + (lane&15); k = kt*32 + kmap(lane>>4, j)  (ISA 16-bit A/B layout)
__global__ void pack_w_kernel(const float* __restrict__ W,
                              __bf16* __restrict__ packed,
                              int Ntiles, int K) {
  size_t idx = (size_t)blockIdx.x * blockDim.x + threadIdx.x;
  size_t total = (size_t)Ntiles * 16 * K;
  if (idx >= total) return;
  int j    = (int)(idx & 15);
  size_t r = idx >> 4;
  int lane = (int)(r & 31); r >>= 5;
  int KT   = K >> 5;
  int kt   = (int)(r % KT);
  int nt   = (int)(r / KT);
  int n  = nt * 16 + (lane & 15);
  int hi = lane >> 4;
  int k  = kt * 32 + ((j < 8) ? (8 * hi + j) : (16 + 8 * hi + (j - 8)));
  packed[idx] = (__bf16)W[(size_t)n * K + k];
}

// ---- x -> bf16, vectorized 4-wide (b128 read, b64 write)
__global__ void cast_bf16x4_kernel(const float4* __restrict__ src,
                                   v4bf* __restrict__ dst, size_t n4) {
  size_t i = (size_t)blockIdx.x * blockDim.x + threadIdx.x;
  size_t stride = (size_t)gridDim.x * blockDim.x;
  for (; i < n4; i += stride) {
    float4 f = src[i];
    v4bf o;
    o[0] = (__bf16)f.x; o[1] = (__bf16)f.y;
    o[2] = (__bf16)f.z; o[3] = (__bf16)f.w;
    dst[i] = o;
  }
}

__global__ void init_state_kernel(const float* __restrict__ h0,
                                  const float* __restrict__ c0,
                                  __bf16* __restrict__ h_bf0,
                                  float* __restrict__ c_st) {
  int i = blockIdx.x * blockDim.x + threadIdx.x;
  if (i < B_DIM * H_DIM) {
    h_bf0[i] = (__bf16)h0[i];
    c_st[i]  = c0[i];
  }
}

// ---- Phase 1: gates_x[r, g] = sum_k x[r,k]*W_ih[g,k] + b_ih[g] + b_hh[g]
// grid: (G/64 = 64 n-groups, SB/64 = 512 m-groups); 1 wave/block computing a
// 64x64 output tile: 4 A frags x 4 B frags -> 16 WMMAs per k-step.
// Output in tile order: gx[(mrow_tile*NT_G + nt)*256 + lane*8 + r].
__global__ __launch_bounds__(32) void gates_x_gemm_kernel(
    const __bf16* __restrict__ x_bf, const __bf16* __restrict__ wih_p,
    const float* __restrict__ b_ih, const float* __restrict__ b_hh,
    float* __restrict__ gx) {
  const int lane = threadIdx.x;
  const int col  = lane & 15;
  const int hi   = lane >> 4;
  const int ng   = blockIdx.x;   // group of 4 n-tiles over G
  const int mg   = blockIdx.y;   // group of 4 m-tiles over SB (== time step)

  v8f acc[4][4];                 // [m][n]
#pragma unroll
  for (int n = 0; n < 4; ++n) {
    const int g = (ng * 4 + n) * 16 + col;
    const float bias = b_ih[g] + b_hh[g];
#pragma unroll
    for (int m = 0; m < 4; ++m)
#pragma unroll
      for (int r = 0; r < 8; ++r) acc[m][n][r] = bias;
  }

#pragma unroll 1
  for (int kt = 0; kt < KT_I; ++kt) {
    const int kbase = kt * 32 + 8 * hi;
    AFrag a[4];
#pragma unroll
    for (int m = 0; m < 4; ++m) {
      const int rowA = mg * 64 + m * 16 + col;  // A-matrix row = lane&15
      const __bf16* ap = x_bf + (size_t)rowA * I_DIM + kbase;
      a[m].h[0] = *(const v8bf*)(ap);
      a[m].h[1] = *(const v8bf*)(ap + 16);
    }
#pragma unroll
    for (int n = 0; n < 4; ++n) {
      const int nt = ng * 4 + n;
      const __bf16* bptr = wih_p + (size_t)(nt * KT_I + kt) * 512 + lane * 16;
      v16bf bf = *(const v16bf*)bptr;
      __builtin_prefetch(bptr + 512, 0, 3);   // next k-tile frag
#pragma unroll
      for (int m = 0; m < 4; ++m)
        acc[m][n] = __builtin_amdgcn_wmma_f32_16x16x32_bf16(
            false, a[m].v, false, bf, (short)0, acc[m][n], false, false);
    }
  }

  // Store in C-fragment tile order: one contiguous 32B v8f per tile per lane.
#pragma unroll
  for (int m = 0; m < 4; ++m) {
    const size_t mrow = (size_t)(mg * 4 + m);
#pragma unroll
    for (int n = 0; n < 4; ++n) {
      float* dst = gx + (mrow * NT_G + (ng * 4 + n)) * 256 + lane * 8;
      *(v8f*)dst = acc[m][n];
    }
  }
}

// ---- Phase 2: one time step. grid: (H/16 = 64 n-tiles, B/16 = 4 m-tiles),
// 1 wave/block. Each wave computes the i,f,g,o 16x16 tiles for its (b,h)
// tile (columns nt, nt+H/16, nt+2H/16, nt+3H/16 of the gate matrix), then
// performs the cell update locally. gates_x tile (fragment-tiled layout) is
// loaded directly as the WMMA C operand with a single v8f load per gate.
__global__ __launch_bounds__(32) void lstm_step_kernel(
    const float* __restrict__ gx_t,      // gates_x + t*B*G (tile order)
    const __bf16* __restrict__ whh_p,
    const __bf16* __restrict__ h_in,     // [B,H] bf16 (parity t)
    __bf16* __restrict__ h_out,          // [B,H] bf16 (parity t+1)
    float* __restrict__ c_st,            // [B,H] fp32 cell state
    float* __restrict__ out_t,           // d_out + t*B*H
    float* __restrict__ h_last, float* __restrict__ c_last, int is_last) {
  const int lane = threadIdx.x;
  const int col  = lane & 15;
  const int hi   = lane >> 4;
  const int nt   = blockIdx.x;   // tile over H
  const int mt   = blockIdx.y;   // tile over B

  v8f acc[4];
#pragma unroll
  for (int q = 0; q < 4; ++q) {
    const int ntg = q * (H_DIM / 16) + nt;   // gate column tile in G
    acc[q] = *(const v8f*)(gx_t + ((size_t)mt * NT_G + ntg) * 256 + lane * 8);
  }

#pragma unroll 1
  for (int kt = 0; kt < KT_H; ++kt) {
    const int rowA = mt * 16 + col;
    const __bf16* ap = h_in + (size_t)rowA * H_DIM + kt * 32 + 8 * hi;
    AFrag a;
    a.h[0] = *(const v8bf*)(ap);
    a.h[1] = *(const v8bf*)(ap + 16);
#pragma unroll
    for (int q = 0; q < 4; ++q) {
      const int ntg = q * (H_DIM / 16) + nt;
      const __bf16* bptr = whh_p + (size_t)(ntg * KT_H + kt) * 512 + lane * 16;
      v16bf bf = *(const v16bf*)bptr;
      __builtin_prefetch(bptr + 512, 0, 3);
      acc[q] = __builtin_amdgcn_wmma_f32_16x16x32_bf16(
          false, a.v, false, bf, (short)0, acc[q], false, false);
    }
  }

#pragma unroll
  for (int r = 0; r < 8; ++r) {
    const int b = mt * 16 + r + 8 * hi;
    const int n = nt * 16 + col;
    const size_t idx = (size_t)b * H_DIM + n;
    const float iv = sigf(acc[0][r]);
    const float fv = sigf(acc[1][r]);
    const float gv = tanhfast(acc[2][r]);
    const float ov = sigf(acc[3][r]);
    const float cold = c_st[idx];
    const float cnew = fv * cold + iv * gv;
    const float hnew = ov * tanhfast(cnew);
    c_st[idx]  = cnew;
    out_t[idx] = hnew;
    h_out[idx] = (__bf16)hnew;
    if (is_last) { h_last[idx] = hnew; c_last[idx] = cnew; }
  }
}

// ---------------------------------------------------------------------------
extern "C" void kernel_launch(void* const* d_in, const int* in_sizes, int n_in,
                              void* d_out, int out_size, void* d_ws,
                              size_t ws_size, hipStream_t stream) {
  (void)in_sizes; (void)n_in; (void)out_size; (void)ws_size;
  const float* x    = (const float*)d_in[0];
  const float* h0   = (const float*)d_in[1];
  const float* c0   = (const float*)d_in[2];
  const float* W_ih = (const float*)d_in[3];
  const float* b_ih = (const float*)d_in[4];
  const float* W_hh = (const float*)d_in[5];
  const float* b_hh = (const float*)d_in[6];
  float* out = (float*)d_out;

  char* ws = (char*)d_ws;
  size_t off = 0;
  auto carve = [&](size_t bytes) -> char* {
    char* p = ws + off;
    off += (bytes + 255) & ~(size_t)255;
    return p;
  };
  __bf16* x_bf  = (__bf16*)carve((size_t)SB * I_DIM * 2);       // 64 MB
  __bf16* wih_p = (__bf16*)carve((size_t)G_DIM * I_DIM * 2);    // 8 MB
  __bf16* whh_p = (__bf16*)carve((size_t)G_DIM * H_DIM * 2);    // 8 MB
  __bf16* h_bf  = (__bf16*)carve((size_t)2 * B_DIM * H_DIM * 2);
  float*  c_st  = (float*)carve((size_t)B_DIM * H_DIM * 4);
  float*  gx    = (float*)carve((size_t)SB * G_DIM * 4);        // 512 MB

  {
    size_t total = (size_t)G_DIM * I_DIM;
    int blocks = (int)((total + 255) / 256);
    pack_w_kernel<<<blocks, 256, 0, stream>>>(W_ih, wih_p, G_DIM / 16, I_DIM);
    pack_w_kernel<<<blocks, 256, 0, stream>>>(W_hh, whh_p, G_DIM / 16, H_DIM);
  }
  cast_bf16x4_kernel<<<4096, 256, 0, stream>>>(
      (const float4*)x, (v4bf*)x_bf, (size_t)SB * I_DIM / 4);
  init_state_kernel<<<(B_DIM * H_DIM + 255) / 256, 256, 0, stream>>>(
      h0, c0, h_bf, c_st);

  gates_x_gemm_kernel<<<dim3(G_DIM / 64, SB / 64), 32, 0, stream>>>(
      x_bf, wih_p, b_ih, b_hh, gx);

  float* h_last = out + (size_t)S_LEN * B_DIM * H_DIM;
  float* c_last = h_last + (size_t)B_DIM * H_DIM;
  for (int t = 0; t < S_LEN; ++t) {
    const float* gx_t = gx + (size_t)t * B_DIM * G_DIM;
    const __bf16* h_in = h_bf + (size_t)(t & 1) * B_DIM * H_DIM;
    __bf16* h_out_p    = h_bf + (size_t)((t + 1) & 1) * B_DIM * H_DIM;
    float* out_t = out + (size_t)t * B_DIM * H_DIM;
    lstm_step_kernel<<<dim3(H_DIM / 16, B_DIM / 16), 32, 0, stream>>>(
        gx_t, whh_p, h_in, h_out_p, c_st, out_t, h_last, c_last,
        (t == S_LEN - 1) ? 1 : 0);
  }
}
